// SCNLayer_17815524344015
// MI455X (gfx1250) — matrix operand order repaired and checked by reading, not compiled, not run on hardware
//
#include <hip/hip_runtime.h>
#include <hip/hip_bf16.h>

// Problem constants (match the reference)
#define NNODES 50000
#define NEDGES 800000
#define DFEAT  128
#define KORD   4
#define DOUT   128
#define KDIM   (KORD * DFEAT)   // 512

typedef __attribute__((ext_vector_type(16))) __bf16 v16bf;
typedef __attribute__((ext_vector_type(8)))  float  v8f;

// ---------------------------------------------------------------------------
// Elementwise helpers (float4-wide for bandwidth)
// ---------------------------------------------------------------------------
__global__ void fill_zero_f4(float4* __restrict__ p, int n4) {
    int i = blockIdx.x * blockDim.x + threadIdx.x;
    if (i < n4) p[i] = make_float4(0.f, 0.f, 0.f, 0.f);
}

// y = -a  (seed for fused Chebyshev step: T_next starts at -T_{i-2})
__global__ void init_neg_f4(float4* __restrict__ y, const float4* __restrict__ a, int n4) {
    int i = blockIdx.x * blockDim.x + threadIdx.x;
    if (i < n4) {
        float4 v = a[i];
        y[i] = make_float4(-v.x, -v.y, -v.z, -v.w);
    }
}

// ---------------------------------------------------------------------------
// SpMM: Y[rows[e]] += scale * vals[e] * X[cols[e]]   (one wave32 per edge)
// 32 lanes x float4 = 128 features: coalesced 512B gather + 512B atomic burst.
// ---------------------------------------------------------------------------
__global__ void spmm_atomic(const int*   __restrict__ rows,
                            const int*   __restrict__ cols,
                            const float* __restrict__ vals,
                            float scale,
                            const float* __restrict__ X,
                            float*       __restrict__ Y,
                            int E) {
    int gid  = blockIdx.x * blockDim.x + threadIdx.x;
    int e    = gid >> 5;          // wave index == edge index
    int lane = gid & 31;
    if (e >= E) return;

    int   r = rows[e];
    int   c = cols[e];
    float v = vals[e] * scale;

    float4 xv = ((const float4*)(X + (size_t)c * DFEAT))[lane];
    float* y  = Y + (size_t)r * DFEAT + lane * 4;
    atomicAdd(y + 0, v * xv.x);
    atomicAdd(y + 1, v * xv.y);
    atomicAdd(y + 2, v * xv.z);
    atomicAdd(y + 3, v * xv.w);
}

// ---------------------------------------------------------------------------
// Pre-swizzle W [KDIM x DOUT] f32 -> bf16 B-fragments.
// Fragment (kt, nt): 32 lanes x 16 bf16, laid out per the CDNA5 16-bit B
// layout: lane L -> n = nt*16 + (L%16); element e -> k = kt*32 + (e&7) + 8*(L>=16) + 16*(e>=8).
// Stored contiguously so each B fragment is one 32-byte per-lane load.
// ---------------------------------------------------------------------------
__global__ void swizzle_W(const float* __restrict__ W, __bf16* __restrict__ Wsw) {
    int idx  = blockIdx.x * blockDim.x + threadIdx.x;   // KDIM*DOUT = 65536
    int elem = idx & 15;
    int lane = (idx >> 4) & 31;
    int t    = idx >> 9;          // kt*8 + nt
    int nt   = t & 7;
    int kt   = t >> 3;
    int n    = nt * 16 + (lane & 15);
    int k    = kt * 32 + (elem & 7) + ((lane >> 4) << 3) + ((elem >> 3) << 4);
    Wsw[idx] = (__bf16)W[k * DOUT + n];
}

// ---------------------------------------------------------------------------
// out = [x | T1 | T2 | T3] @ W + b  via V_WMMA_F32_16X16X32_BF16.
// 256 threads = 8 waves; each wave: 16 rows x 128 cols (8 accumulator tiles).
//
// The full pre-swizzled W (128 KB bf16) is staged in LDS once per block, so
// all B fragments come from ds_load_b128 and the vmem pipe carries only the
// streaming A loads. A fragments are built on the fly (f32 -> bf16) per the
// ISA 16-bit A layout: lane L holds row M = L%16; elements 0..7 ->
// K = koff..koff+7, elements 8..15 -> K = koff+16..koff+23, koff = 8*(L>=16).
// ---------------------------------------------------------------------------
__global__ void __launch_bounds__(256)
cheb_gemm_wmma(const float*  __restrict__ X0,
               const float*  __restrict__ T1,
               const float*  __restrict__ T2,
               const float*  __restrict__ T3,
               const __bf16* __restrict__ Wsw,
               const float*  __restrict__ bias,
               float*        __restrict__ out) {
    // Whole W in fragment layout: 512*128 bf16 = 128 KB (<= 320 KB WGP LDS).
    __shared__ __bf16 sW[KDIM * DOUT];

    // Cooperative fill: 131072 B = 8192 x uint4; 32 iterations per thread.
    {
        const uint4* src = (const uint4*)Wsw;
        uint4*       dst = (uint4*)sW;
        #pragma unroll
        for (int i = 0; i < (KDIM * DOUT * 2) / (16 * 256); ++i)
            dst[i * 256 + threadIdx.x] = src[i * 256 + threadIdx.x];
    }
    __syncthreads();

    const float* seg[4] = {X0, T1, T2, T3};

    int lane  = threadIdx.x & 31;
    int wave  = threadIdx.x >> 5;
    int mbase = blockIdx.x * 128 + wave * 16;

    int mrow = mbase + (lane & 15);
    int mc   = mrow < NNODES ? mrow : NNODES - 1;   // clamp; store is guarded
    int koff = (lane >> 4) * 8;

    v8f acc[8] = {};
    const v16bf* Wf = (const v16bf*)sW;

    #pragma unroll
    for (int s = 0; s < 4; ++s) {
        const float* arow = seg[s] + (size_t)mc * DFEAT;
        #pragma unroll
        for (int kk = 0; kk < DFEAT; kk += 32) {
            // Build bf16 A fragment: two contiguous 8-float runs per lane.
            v16bf a;
            #pragma unroll
            for (int e = 0; e < 8; ++e) {
                a[e]     = (__bf16)arow[kk + koff + e];
                a[e + 8] = (__bf16)arow[kk + koff + 16 + e];
            }
            int kt = s * 4 + (kk >> 5);
            #pragma unroll
            for (int nt = 0; nt < 8; ++nt) {
                v16bf b = Wf[(kt * 8 + nt) * 32 + lane];
                acc[nt] = __builtin_amdgcn_wmma_f32_16x16x32_bf16(
                    /*neg_a=*/false, a, /*neg_b=*/false, b,
                    /*c_mod=*/(short)0, acc[nt],
                    /*reuse_a=*/false, /*reuse_b=*/false);
            }
        }
    }

    // C/D layout: lane L -> n = L%16; VGPR r -> m = r + 8*(L>=16).
    int n0  = lane & 15;
    int mhi = (lane >> 4) * 8;
    #pragma unroll
    for (int nt = 0; nt < 8; ++nt) {
        int   n  = nt * 16 + n0;
        float bv = bias[n];
        #pragma unroll
        for (int r = 0; r < 8; ++r) {
            int m = mbase + mhi + r;
            if (m < NNODES) out[(size_t)m * DOUT + n] = acc[nt][r] + bv;
        }
    }
}

// ---------------------------------------------------------------------------
// Host-side launch. Workspace layout (floats): T1 | T2 | T3 | Wsw(bf16).
// ---------------------------------------------------------------------------
extern "C" void kernel_launch(void* const* d_in, const int* in_sizes, int n_in,
                              void* d_out, int out_size, void* d_ws, size_t ws_size,
                              hipStream_t stream) {
    const float* x    = (const float*)d_in[0];
    const float* vals = (const float*)d_in[1];
    const float* W    = (const float*)d_in[2];
    const float* b    = (const float*)d_in[3];
    const int*   rows = (const int*)d_in[4];
    const int*   cols = (const int*)d_in[5];
    float* out = (float*)d_out;

    const size_t NF = (size_t)NNODES * DFEAT;      // 6.4M floats per term
    float*  T1  = (float*)d_ws;
    float*  T2  = T1 + NF;
    float*  T3  = T2 + NF;
    __bf16* Wsw = (__bf16*)(T3 + NF);              // 64K bf16 = 128KB

    const int tb    = 256;
    const int n4    = (int)(NF / 4);
    const int gInit = (n4 + tb - 1) / tb;
    const int gSpmm = (NEDGES * 32 + tb - 1) / tb; // one wave32 per edge
    const int gSwiz = (KDIM * DOUT) / tb;          // 65536 / 256
    const int gGemm = (NNODES + 127) / 128;

    // T1 = L @ x
    fill_zero_f4<<<gInit, tb, 0, stream>>>((float4*)T1, n4);
    spmm_atomic<<<gSpmm, tb, 0, stream>>>(rows, cols, vals, 1.0f, x, T1, NEDGES);

    // T2 = 2 L T1 - x   (seed -x, atomically add 2*val*T1[col])
    init_neg_f4<<<gInit, tb, 0, stream>>>((float4*)T2, (const float4*)x, n4);
    spmm_atomic<<<gSpmm, tb, 0, stream>>>(rows, cols, vals, 2.0f, T1, T2, NEDGES);

    // T3 = 2 L T2 - T1
    init_neg_f4<<<gInit, tb, 0, stream>>>((float4*)T3, (const float4*)T1, n4);
    spmm_atomic<<<gSpmm, tb, 0, stream>>>(rows, cols, vals, 2.0f, T2, T3, NEDGES);

    // W -> bf16 WMMA B-fragment layout
    swizzle_W<<<gSwiz, tb, 0, stream>>>(W, Wsw);

    // out = [x|T1|T2|T3] @ W + b
    cheb_gemm_wmma<<<gGemm, tb, 0, stream>>>(x, T1, T2, T3, Wsw, b, out);
}
